// AverageAttention_50053548868205
// MI455X (gfx1250) — matrix operand order repaired and checked by reading, not compiled, not run on hardware
//
#include <hip/hip_runtime.h>
#include <hip/hip_bf16.h>

typedef __bf16 bf16;
typedef __attribute__((ext_vector_type(16))) __bf16 v16bf;
typedef __attribute__((ext_vector_type(8)))  __bf16 v8bf;
typedef __attribute__((ext_vector_type(8)))  float  v8f;
typedef int i4v __attribute__((vector_size(16)));

#define D_DIM 1024
#define L_DIM 4096
#define B_DIM 8
#define M_DIM (B_DIM * L_DIM)      // 32768
#define LDS_STRIDE 40              // bf16 elems; 80B rows -> conflict-free 16B lane reads

#if __has_builtin(__builtin_amdgcn_global_load_async_to_lds_b128)
#define HAVE_ASYNC_LDS 1
#define AS1 __attribute__((address_space(1)))
#define AS3 __attribute__((address_space(3)))
#else
#define HAVE_ASYNC_LDS 0
#endif

__device__ __forceinline__ bf16 f2bf(float f) {
    unsigned u = __builtin_bit_cast(unsigned, f);
    unsigned r = u + 0x7FFFu + ((u >> 16) & 1u);   // round-to-nearest-even
    unsigned short s = (unsigned short)(r >> 16);
    return __builtin_bit_cast(bf16, s);
}

__device__ __forceinline__ float sigmoidf(float x) {
    return 1.0f / (1.0f + __expf(-x));
}

// stage 16 bf16 (32B) global -> LDS; async DMA path if available
__device__ __forceinline__ void stage_a16(const bf16* g, bf16* l) {
#if HAVE_ASYNC_LDS
    AS1 i4v* gp = (AS1 i4v*)(AS1 void*)g;
    AS3 i4v* lp = (AS3 i4v*)(AS3 void*)l;
    __builtin_amdgcn_global_load_async_to_lds_b128(gp, lp, 0, 0);
    __builtin_amdgcn_global_load_async_to_lds_b128(gp + 1, lp + 1, 0, 0);
#else
    ((uint4*)l)[0] = ((const uint4*)g)[0];
    ((uint4*)l)[1] = ((const uint4*)g)[1];
#endif
}

__device__ __forceinline__ void async_wait0() {
#if HAVE_ASYNC_LDS
#if __has_builtin(__builtin_amdgcn_s_wait_asynccnt)
    __builtin_amdgcn_s_wait_asynccnt(0);
#else
    asm volatile("s_wait_asynccnt 0" ::: "memory");
#endif
#endif
}

// pack 2 K-rows (8 n each) into transposed tile Bt[n][k]; k contiguous -> b32 stores
__device__ __forceinline__ void store_bt(bf16* base, int nOff, int kr, uint4 r0, uint4 r1) {
    const unsigned short* p0 = (const unsigned short*)&r0;
    const unsigned short* p1 = (const unsigned short*)&r1;
#pragma unroll
    for (int e = 0; e < 8; ++e) {
        unsigned v = (unsigned)p0[e] | ((unsigned)p1[e] << 16);
        *(unsigned*)(base + (size_t)(nOff + e) * LDS_STRIDE + kr) = v;
    }
}

// A fragment (16x32 bf16, ISA 7.12.2): lane m = l&15; half-wave selects K 0-7/16-23 vs 8-15/24-31
__device__ __forceinline__ v16bf load_frag_a(const bf16* tile, int row0, int lane) {
    int r  = row0 + (lane & 15);
    int kb = (lane >> 4) * 8;
    const bf16* p = tile + r * LDS_STRIDE + kb;
    v8bf lo = *(const v8bf*)(p);
    v8bf hi = *(const v8bf*)(p + 16);
    return __builtin_shufflevector(lo, hi, 0,1,2,3,4,5,6,7,8,9,10,11,12,13,14,15);
}

// B fragment (32x16 bf16) from transposed tile Bt[n][k]: lane n = l&15; half-wave selects K 0-7/16-23 vs 8-15/24-31
__device__ __forceinline__ v16bf load_frag_b(const bf16* tile, int row0, int lane) {
    int r  = row0 + (lane & 15);
    int kb = (lane >> 4) * 8;
    const bf16* p = tile + r * LDS_STRIDE + kb;
    v8bf lo = *(const v8bf*)(p);
    v8bf hi = *(const v8bf*)(p + 16);
    return __builtin_shufflevector(lo, hi, 0,1,2,3,4,5,6,7,8,9,10,11,12,13,14,15);
}

// ---------------- Kernel 1: running mean over L + bf16 copy of inputs ----------------
__global__ __launch_bounds__(256) void cumavg_kernel(const float* __restrict__ x,
                                                     float* __restrict__ avg,
                                                     bf16* __restrict__ xbf) {
    int c = blockIdx.x * 256 + threadIdx.x;      // 0 .. B*D-1
    int b = c / D_DIM, d = c - b * D_DIM;
    size_t gbase = (size_t)b * L_DIM * D_DIM + d;
    const float* xp = x + gbase;
    float acc = 0.0f;
    for (int l = 0; l < L_DIM; l += 4) {
        float v0 = xp[(size_t)(l + 0) * D_DIM];
        float v1 = xp[(size_t)(l + 1) * D_DIM];
        float v2 = xp[(size_t)(l + 2) * D_DIM];
        float v3 = xp[(size_t)(l + 3) * D_DIM];
        float a0 = acc + v0, a1 = a0 + v1, a2 = a1 + v2, a3 = a2 + v3;
        avg[gbase + (size_t)(l + 0) * D_DIM] = a0 * __builtin_amdgcn_rcpf((float)(l + 1));
        avg[gbase + (size_t)(l + 1) * D_DIM] = a1 * __builtin_amdgcn_rcpf((float)(l + 2));
        avg[gbase + (size_t)(l + 2) * D_DIM] = a2 * __builtin_amdgcn_rcpf((float)(l + 3));
        avg[gbase + (size_t)(l + 3) * D_DIM] = a3 * __builtin_amdgcn_rcpf((float)(l + 4));
        xbf[gbase + (size_t)(l + 0) * D_DIM] = f2bf(v0);
        xbf[gbase + (size_t)(l + 1) * D_DIM] = f2bf(v1);
        xbf[gbase + (size_t)(l + 2) * D_DIM] = f2bf(v2);
        xbf[gbase + (size_t)(l + 3) * D_DIM] = f2bf(v3);
        acc = a3;
    }
}

// ---------------- Kernel 2: LayerNorm over D (one wave per row) ----------------
__global__ __launch_bounds__(256) void ln_kernel(const float* __restrict__ avg,
                                                 const float* __restrict__ g,
                                                 const float* __restrict__ bta,
                                                 bf16* __restrict__ h) {
    int wave = threadIdx.x >> 5, lane = threadIdx.x & 31;
    size_t row = (size_t)blockIdx.x * 8 + wave;
    const float* p = avg + row * D_DIM;
    float v[32];
    float s = 0.0f;
#pragma unroll
    for (int i = 0; i < 32; ++i) { v[i] = p[i * 32 + lane]; s += v[i]; }
#pragma unroll
    for (int o = 16; o > 0; o >>= 1) s += __shfl_xor(s, o, 32);
    float mu = s * (1.0f / (float)D_DIM);
    float vs = 0.0f;
#pragma unroll
    for (int i = 0; i < 32; ++i) { float dd = v[i] - mu; vs += dd * dd; }
#pragma unroll
    for (int o = 16; o > 0; o >>= 1) vs += __shfl_xor(vs, o, 32);
    float inv = rsqrtf(vs * (1.0f / (float)D_DIM) + 1e-6f);
    bf16* hp = h + row * D_DIM;
#pragma unroll
    for (int i = 0; i < 32; ++i) {
        int col = i * 32 + lane;
        hp[col] = f2bf((v[i] - mu) * inv * g[col] + bta[col]);
    }
}

// ---------------- Kernel 3: fp32 -> bf16 weight conversion ----------------
__global__ __launch_bounds__(256) void cvt_kernel(const float* __restrict__ src,
                                                  bf16* __restrict__ dst, int n) {
    int i = blockIdx.x * 256 + threadIdx.x;
    if (i < n) dst[i] = f2bf(src[i]);
}

// ---------------- GEMM (128x128 tile, BK=32, double-buffered, bf16 WMMA, f32 acc) ----------------
// EPI 0: out_bf = relu(acc + bias)
// EPI 1: v = acc + bias + addsrc; out_f = v; out_bf = v
template <int EPI>
__global__ __launch_bounds__(256) void gemm_kernel(const bf16* __restrict__ A,
                                                   const bf16* __restrict__ B,
                                                   const float* __restrict__ bias,
                                                   const float* __restrict__ addsrc,
                                                   bf16* __restrict__ out_bf,
                                                   float* __restrict__ out_f,
                                                   int M, int N, int K) {
    __shared__ bf16 As[2][128 * LDS_STRIDE];
    __shared__ bf16 Bt[2][128 * LDS_STRIDE];

    int tid = threadIdx.x;
    int n0 = blockIdx.x * 128;
    int m0 = blockIdx.y * 128;
    int lane = tid & 31, wave = tid >> 5;
    int wm = (wave >> 1) * 32;   // 0,32,64,96
    int wn = (wave & 1) * 64;    // 0,64

    // staging coords
    int arow = tid >> 1, akh = (tid & 1) * 16;
    int bnOff = (tid & 15) * 8;
    int bkr  = (tid >> 4) * 2;
    const bf16* aSrc = A + (size_t)(m0 + arow) * K + akh;
    bf16* aDst0 = &As[0][arow * LDS_STRIDE + akh];
    bf16* aDst1 = &As[1][arow * LDS_STRIDE + akh];

    v8f acc[2][4] = {};

    // prologue: stage tile 0
    stage_a16(aSrc, aDst0);
    {
        const bf16* src = B + (size_t)bkr * N + n0 + bnOff;
        uint4 r0 = *(const uint4*)src;
        uint4 r1 = *(const uint4*)(src + N);
        store_bt(&Bt[0][0], bnOff, bkr, r0, r1);
    }
    async_wait0();
    __syncthreads();

    int nk = K >> 5;
    int buf = 0;
    for (int kt = 0; kt < nk; ++kt) {
        uint4 r0, r1;
        bool more = (kt + 1) < nk;
        if (more) {
            int k0n = (kt + 1) << 5;
            stage_a16(aSrc + k0n, buf ? aDst0 : aDst1);
            const bf16* src = B + (size_t)(k0n + bkr) * N + n0 + bnOff;
            r0 = *(const uint4*)src;
            r1 = *(const uint4*)(src + N);
        }
        // compute from current buffer
        v16bf af[2], bfg[4];
#pragma unroll
        for (int i = 0; i < 2; ++i) af[i] = load_frag_a(As[buf], wm + i * 16, lane);
#pragma unroll
        for (int j = 0; j < 4; ++j) bfg[j] = load_frag_b(Bt[buf], wn + j * 16, lane);
#pragma unroll
        for (int i = 0; i < 2; ++i)
#pragma unroll
            for (int j = 0; j < 4; ++j)
                acc[i][j] = __builtin_amdgcn_wmma_f32_16x16x32_bf16(
                    false, af[i], false, bfg[j], (short)0, acc[i][j], false, false);
        if (more) {
            store_bt(&Bt[buf ^ 1][0], bnOff, bkr, r0, r1);
            async_wait0();
        }
        __syncthreads();
        buf ^= 1;
    }

    // epilogue: lane l holds (m = base + r + 8*(l>>4), n = base + (l&15)) per acc vgpr r
    int mrow0 = m0 + wm + 8 * (lane >> 4);
    int ncol0 = n0 + wn + (lane & 15);
#pragma unroll
    for (int i = 0; i < 2; ++i) {
#pragma unroll
        for (int j = 0; j < 4; ++j) {
            int n = ncol0 + j * 16;
            float bv = bias[n];
#pragma unroll
            for (int r = 0; r < 8; ++r) {
                int m = mrow0 + i * 16 + r;
                size_t idx = (size_t)m * N + n;
                float v = acc[i][j][r] + bv;
                if (EPI == 0) {
                    v = v > 0.0f ? v : 0.0f;
                    out_bf[idx] = f2bf(v);
                } else {
                    v += addsrc[idx];
                    out_f[idx] = v;
                    out_bf[idx] = f2bf(v);
                }
            }
        }
    }
}

// ---------------- GEMM3: dual-gate GEMM + fused sigmoid gating (double-buffered) ----------------
__global__ __launch_bounds__(256) void gemm3_kernel(const bf16* __restrict__ Xbf,
                                                    const bf16* __restrict__ AObf,
                                                    const bf16* __restrict__ WG,  // 2048x2048 bf16
                                                    const float* __restrict__ bg,
                                                    const float* __restrict__ Xf,
                                                    const float* __restrict__ AOf,
                                                    float* __restrict__ out) {
    __shared__ bf16 As[2][128 * LDS_STRIDE];
    __shared__ bf16 Bti[2][64 * LDS_STRIDE];
    __shared__ bf16 Btf[2][64 * LDS_STRIDE];

    const int KW = 2 * D_DIM;
    int tid = threadIdx.x;
    int n0 = blockIdx.x * 64;
    int m0 = blockIdx.y * 128;
    int lane = tid & 31, wave = tid >> 5;
    int wm = (wave >> 1) * 32;    // 0,32,64,96
    int wn = (wave & 1) * 32;     // 0,32

    // A staging coords
    int arow = tid >> 1, akh = (tid & 1) * 16;
    const bf16* aX  = Xbf  + (size_t)(m0 + arow) * D_DIM + akh;
    const bf16* aAO = AObf + (size_t)(m0 + arow) * D_DIM + akh;
    // B staging: threads 0-127 -> gate-i tile, 128-255 -> gate-f tile; 8n x 2k each
    int sel  = tid >> 7;
    int tt   = tid & 127;
    int bnOff = (tt & 7) * 8;
    int bkr   = (tt >> 3) * 2;
    const bf16* bSrc = WG + n0 + (size_t)sel * D_DIM + bnOff;

    v8f acci[2][2] = {};
    v8f accf[2][2] = {};

    // prologue: stage tile 0 (k0 = 0 -> A from Xbf)
    stage_a16(aX, &As[0][arow * LDS_STRIDE + akh]);
    {
        const bf16* src = bSrc + (size_t)bkr * KW;
        uint4 r0 = *(const uint4*)src;
        uint4 r1 = *(const uint4*)(src + KW);
        store_bt(sel ? &Btf[0][0] : &Bti[0][0], bnOff, bkr, r0, r1);
    }
    async_wait0();
    __syncthreads();

    const int nk = (2 * D_DIM) >> 5;   // 64
    int buf = 0;
    for (int kt = 0; kt < nk; ++kt) {
        uint4 r0, r1;
        bool more = (kt + 1) < nk;
        if (more) {
            int k0n = (kt + 1) << 5;
            const bf16* ag = (k0n < D_DIM) ? (aX + k0n) : (aAO + (k0n - D_DIM));
            stage_a16(ag, &As[buf ^ 1][arow * LDS_STRIDE + akh]);
            const bf16* src = bSrc + (size_t)(k0n + bkr) * KW;
            r0 = *(const uint4*)src;
            r1 = *(const uint4*)(src + KW);
        }
        v16bf af[2], bi[2], bff[2];
#pragma unroll
        for (int i = 0; i < 2; ++i) af[i] = load_frag_a(As[buf], wm + i * 16, lane);
#pragma unroll
        for (int j = 0; j < 2; ++j) {
            bi[j]  = load_frag_b(Bti[buf], wn + j * 16, lane);
            bff[j] = load_frag_b(Btf[buf], wn + j * 16, lane);
        }
#pragma unroll
        for (int i = 0; i < 2; ++i)
#pragma unroll
            for (int j = 0; j < 2; ++j) {
                acci[i][j] = __builtin_amdgcn_wmma_f32_16x16x32_bf16(
                    false, af[i], false, bi[j], (short)0, acci[i][j], false, false);
                accf[i][j] = __builtin_amdgcn_wmma_f32_16x16x32_bf16(
                    false, af[i], false, bff[j], (short)0, accf[i][j], false, false);
            }
        if (more) {
            store_bt(sel ? &Btf[buf ^ 1][0] : &Bti[buf ^ 1][0], bnOff, bkr, r0, r1);
            async_wait0();
        }
        __syncthreads();
        buf ^= 1;
    }

    int mrow0 = m0 + wm + 8 * (lane >> 4);
    int ncol0 = n0 + wn + (lane & 15);
#pragma unroll
    for (int i = 0; i < 2; ++i) {
#pragma unroll
        for (int j = 0; j < 2; ++j) {
            int n = ncol0 + j * 16;
            float bgi = bg[n];
            float bgf = bg[n + D_DIM];
#pragma unroll
            for (int r = 0; r < 8; ++r) {
                int m = mrow0 + i * 16 + r;
                size_t idx = (size_t)m * D_DIM + n;
                float gi = sigmoidf(acci[i][j][r] + bgi);
                float gf = sigmoidf(accf[i][j][r] + bgf);
                out[idx] = gi * Xf[idx] + gf * AOf[idx];
            }
        }
    }
}

extern "C" void kernel_launch(void* const* d_in, const int* in_sizes, int n_in,
                              void* d_out, int out_size, void* d_ws, size_t ws_size,
                              hipStream_t stream) {
    const float* inputs = (const float*)d_in[0];
    const float* w1 = (const float*)d_in[1];
    const float* b1 = (const float*)d_in[2];
    const float* w2 = (const float*)d_in[3];
    const float* b2 = (const float*)d_in[4];
    const float* ln_g = (const float*)d_in[5];
    const float* ln_b = (const float*)d_in[6];
    const float* wg = (const float*)d_in[7];
    const float* bg = (const float*)d_in[8];

    char* ws = (char*)d_ws;
    const size_t SZ_F32 = (size_t)M_DIM * D_DIM * 4;   // 128 MB
    const size_t SZ_BF  = (size_t)M_DIM * D_DIM * 2;   // 64 MB
    float* avg     = (float*)(ws);
    float* avgout  = (float*)(ws + SZ_F32);
    bf16*  xbf     = (bf16*)(ws + 2 * SZ_F32);
    bf16*  hbf     = (bf16*)(ws + 2 * SZ_F32 + SZ_BF);
    bf16*  interbf = (bf16*)(ws + 2 * SZ_F32 + 2 * SZ_BF);
    bf16*  aobf    = (bf16*)(ws + 2 * SZ_F32 + 3 * SZ_BF);
    bf16*  w1b     = (bf16*)(ws + 2 * SZ_F32 + 4 * SZ_BF);
    bf16*  w2b     = (bf16*)(ws + 2 * SZ_F32 + 4 * SZ_BF + (size_t)D_DIM * D_DIM * 2);
    bf16*  wgb     = (bf16*)(ws + 2 * SZ_F32 + 4 * SZ_BF + (size_t)D_DIM * D_DIM * 4);

    cumavg_kernel<<<(B_DIM * D_DIM) / 256, 256, 0, stream>>>(inputs, avg, xbf);
    ln_kernel<<<M_DIM / 8, 256, 0, stream>>>(avg, ln_g, ln_b, hbf);
    cvt_kernel<<<(D_DIM * D_DIM + 255) / 256, 256, 0, stream>>>(w1, w1b, D_DIM * D_DIM);
    cvt_kernel<<<(D_DIM * D_DIM + 255) / 256, 256, 0, stream>>>(w2, w2b, D_DIM * D_DIM);
    cvt_kernel<<<(4 * D_DIM * D_DIM + 255) / 256, 256, 0, stream>>>(wg, wgb, 4 * D_DIM * D_DIM);
    gemm_kernel<0><<<dim3(D_DIM / 128, M_DIM / 128), 256, 0, stream>>>(
        hbf, w1b, b1, nullptr, interbf, nullptr, M_DIM, D_DIM, D_DIM);
    gemm_kernel<1><<<dim3(D_DIM / 128, M_DIM / 128), 256, 0, stream>>>(
        interbf, w2b, b2, avg, aobf, avgout, M_DIM, D_DIM, D_DIM);
    gemm3_kernel<<<dim3(D_DIM / 64, M_DIM / 128), 256, 0, stream>>>(
        xbf, aobf, wgb, bg, inputs, avgout, (float*)d_out);
}
